// BlockGlobalSelfAttention_2525440770115
// MI455X (gfx1250) — matrix operand order repaired and checked by reading, not compiled, not run on hardware
//
#include <hip/hip_runtime.h>

// ---------------------------------------------------------------------------
// MI455X (gfx1250) block+global self-attention.
// - All matmuls on v_wmma_f32_16x16x32_bf16 (f32 accumulate): the workload is
//   ~230 GFLOP vs ~185MB unique traffic -> compute-bound on the matrix pipe,
//   and bf16 WMMA has 8x the K-throughput of the f32 16x16x4 shape.
// - H and W are converted to bf16 ONCE (W also transposed to [n][k]) so the
//   GEMM inner loop is a pure bf16 copy, staged with
//   global_load_async_to_lds_b128 (ASYNCcnt) and DOUBLE-BUFFERED: panel k+1's
//   copies are issued before computing panel k; s_wait_asynccnt 4 drains only
//   the older panel (async completions are in-order per ISA 4.1).
// - L2 is 192MB: Hbf(32MB)+Wt(8MB) are L2-resident across tile re-reads, so
//   HBM sees only compulsory traffic (~270MB ~= 12us @ 23.3TB/s).
// - Fragment layouts (wave32): A 16x32 and B 32x16 both in contiguous-16
//   k order at half*16 (not the ISA table order), C/D rows r / r+8 per lane half.
// ---------------------------------------------------------------------------

#define B_ 2
#define S_ 4096
#define HID_ 2048
#define NH_ 16
#define HD_ 128
#define BLK_ 128
#define NB_ 32          // S_/BLK_
#define TOPK_ 64
#define SCALE_ 0.0883883476483184f   // 1/sqrt(128)
#define NEGV_ -1000000000.0f

typedef __attribute__((ext_vector_type(16))) __bf16 v16bf;
typedef __attribute__((ext_vector_type(8)))  float  v8f;

static __device__ __forceinline__ __bf16 f2bf(float x) {
    union { float f; unsigned u; } v; v.f = x;
    unsigned r = v.u + 0x7FFFu + ((v.u >> 16) & 1u);   // round-to-nearest-even
    unsigned short h = (unsigned short)(r >> 16);
    __bf16 b; __builtin_memcpy(&b, &h, 2);
    return b;
}

// load 16 bf16 as two 16-byte chunks (A-fragment style: p0 -> elems 0-7, p1 -> 8-15)
static __device__ __forceinline__ v16bf load_frag2(const __bf16* p0, const __bf16* p1) {
    v16bf f;
    ((uint4*)&f)[0] = *(const uint4*)p0;
    ((uint4*)&f)[1] = *(const uint4*)p1;
    return f;
}
// load 16 contiguous bf16 (B-fragment style)
static __device__ __forceinline__ v16bf load_frag1(const __bf16* p) {
    return load_frag2(p, p + 8);
}

static __device__ __forceinline__ v8f wmma_bf16(v16bf a, v16bf b, v8f c) {
    return __builtin_amdgcn_wmma_f32_16x16x32_bf16(false, a, false, b, (short)0, c,
                                                   false, false);
}

static __device__ __forceinline__ float half16_max(float v) {
    v = fmaxf(v, __shfl_xor(v, 1, 32));
    v = fmaxf(v, __shfl_xor(v, 2, 32));
    v = fmaxf(v, __shfl_xor(v, 4, 32));
    v = fmaxf(v, __shfl_xor(v, 8, 32));
    return v;
}
static __device__ __forceinline__ float half16_sum(float v) {
    v += __shfl_xor(v, 1, 32);
    v += __shfl_xor(v, 2, 32);
    v += __shfl_xor(v, 4, 32);
    v += __shfl_xor(v, 8, 32);
    return v;
}

// CDNA5 async global->LDS copy (16B per lane), tracked by ASYNCcnt.
static __device__ __forceinline__ void async_copy16(unsigned lds_off, const void* gaddr) {
    asm volatile("global_load_async_to_lds_b128 %0, %1, off"
                 :: "v"(lds_off), "v"((unsigned long long)(uintptr_t)gaddr)
                 : "memory");
}
static __device__ __forceinline__ void wait_async_le0() {
    asm volatile("s_wait_asynccnt 0x0" ::: "memory");
}
// drain only the older panel's 4 copies (async completions are in-order)
static __device__ __forceinline__ void wait_async_le4() {
    asm volatile("s_wait_asynccnt 0x4" ::: "memory");
}
static __device__ __forceinline__ unsigned lds_addr32(const void* p) {
    return (unsigned)(uintptr_t)p;   // LDS aperture: addr[31:0] is the LDS offset
}

// ---------------------------------------------------------------------------
// Kernel 0a: H f32 -> bf16 (flat, 8 elems/thread).
// ---------------------------------------------------------------------------
__global__ __launch_bounds__(256) void cvt_h(const float* __restrict__ in,
                                             __bf16* __restrict__ out) {
    size_t i = ((size_t)blockIdx.x * 256 + threadIdx.x) * 8;
    float4 a = *(const float4*)(in + i);
    float4 b = *(const float4*)(in + i + 4);
    alignas(16) __bf16 t[8];
    t[0] = f2bf(a.x); t[1] = f2bf(a.y); t[2] = f2bf(a.z); t[3] = f2bf(a.w);
    t[4] = f2bf(b.x); t[5] = f2bf(b.y); t[6] = f2bf(b.z); t[7] = f2bf(b.w);
    *(uint4*)(out + i) = *(uint4*)t;
}

// ---------------------------------------------------------------------------
// Kernel 0b: W[k][n] f32 -> Wt[n][k] bf16 (LDS-tiled transpose + convert).
// ---------------------------------------------------------------------------
__global__ __launch_bounds__(256) void cvt_w(const float* __restrict__ W,
                                             __bf16* __restrict__ Wt) {
    __shared__ __bf16 tile[64 * 72];
    const int tid = threadIdx.x;
    const int kb = blockIdx.y * 64, nb = blockIdx.x * 64;
#pragma unroll
    for (int i = 0; i < 4; ++i) {
        int idx = tid + i * 256;               // 64 k-rows x 16 n-quads
        int kr = idx >> 4, nc = (idx & 15) << 2;
        float4 w = *(const float4*)(W + (size_t)(kb + kr) * HID_ + nb + nc);
        tile[(nc + 0) * 72 + kr] = f2bf(w.x);
        tile[(nc + 1) * 72 + kr] = f2bf(w.y);
        tile[(nc + 2) * 72 + kr] = f2bf(w.z);
        tile[(nc + 3) * 72 + kr] = f2bf(w.w);
    }
    __syncthreads();
#pragma unroll
    for (int i = 0; i < 2; ++i) {
        int idx = tid + i * 256;               // 64 n-rows x 8 k-octets
        int nr = idx >> 3, kc = (idx & 7) << 3;
        *(uint4*)(Wt + (size_t)(nb + nr) * HID_ + kb + kc) = *(uint4*)&tile[nr * 72 + kc];
    }
}

// ---------------------------------------------------------------------------
// Kernel 1: C[8192x2048] = Hbf[8192x2048] @ Wt^T + bias, out bf16.
// 8 waves; block tile 128x128; wave tile 64x32 (4x2 WMMA accs).
// A/B panels staged via global_load_async_to_lds_b128, ping-pong double
// buffered so copies for panel k+1 fly behind panel k's WMMAs.
// ---------------------------------------------------------------------------
__global__ __launch_bounds__(256) void qkv_gemm(const __bf16* __restrict__ Hb,
                                                const __bf16* __restrict__ Wt,
                                                const float* __restrict__ bias,
                                                __bf16* __restrict__ out) {
    __shared__ __bf16 ldsA[2][128 * 40];
    __shared__ __bf16 ldsB[2][128 * 40];   // stored as [n][k]

    const int tid  = threadIdx.x;
    const int wave = tid >> 5;
    const int lane = tid & 31;
    const int hi   = lane >> 4;
    const int l15  = lane & 15;
    const int klo  = hi ? 8 : 0;        // unused; A frags also use klb
    const int klb  = hi ? 16 : 0;       // B-fragment K offset

    const int bm = blockIdx.y * 128;
    const int bn = blockIdx.x * 128;
    const int wm = (wave >> 2) * 64;
    const int wn = (wave & 3) * 32;

    const unsigned aBase = lds_addr32(ldsA[0]);
    const unsigned bBase = lds_addr32(ldsB[0]);
    const unsigned bufStride = 128u * 40u * 2u;   // bytes per buffer

    // per-thread copy chunk coords (rows r0 and r0+64, same 8-elem column)
    const int r0 = tid >> 2;
    const int c0 = (tid & 3) << 3;
    const unsigned d0 = (unsigned)(r0 * 40 + c0) * 2u;
    const unsigned d1 = d0 + 64u * 40u * 2u;

    auto stage = [&](int k0, int buf) {
        unsigned aB = aBase + (unsigned)buf * bufStride;
        unsigned bB = bBase + (unsigned)buf * bufStride;
        async_copy16(aB + d0, Hb + (size_t)(bm + r0) * HID_ + k0 + c0);
        async_copy16(bB + d0, Wt + (size_t)(bn + r0) * HID_ + k0 + c0);
        async_copy16(aB + d1, Hb + (size_t)(bm + r0 + 64) * HID_ + k0 + c0);
        async_copy16(bB + d1, Wt + (size_t)(bn + r0 + 64) * HID_ + k0 + c0);
    };

    v8f acc[4][2];
#pragma unroll
    for (int i = 0; i < 4; ++i)
#pragma unroll
        for (int j = 0; j < 2; ++j) acc[i][j] = (v8f){0,0,0,0,0,0,0,0};

    stage(0, 0);
    for (int it = 0; it < HID_ / 32; ++it) {
        const int buf = it & 1;
        if (it + 1 < HID_ / 32) {
            stage((it + 1) * 32, buf ^ 1);  // prefetch next panel into other buffer
            wait_async_le4();               // drain only current panel's copies
        } else {
            wait_async_le0();
        }
        __syncthreads();                    // all waves' current-panel copies landed

        const __bf16* pA = ldsA[buf];
        const __bf16* pB = ldsB[buf];
        v16bf af[4], bfr[2];
#pragma unroll
        for (int i = 0; i < 4; ++i) {
            const __bf16* p = &pA[(wm + i * 16 + l15) * 40];
            af[i] = load_frag1(p + klb);
        }
#pragma unroll
        for (int j = 0; j < 2; ++j) {
            const __bf16* p = &pB[(wn + j * 16 + l15) * 40];
            bfr[j] = load_frag1(p + klb);
        }
#pragma unroll
        for (int i = 0; i < 4; ++i)
#pragma unroll
            for (int j = 0; j < 2; ++j) acc[i][j] = wmma_bf16(af[i], bfr[j], acc[i][j]);

        __syncthreads();                    // buf free before it+1 stages into it
    }

#pragma unroll
    for (int i = 0; i < 4; ++i)
#pragma unroll
        for (int j = 0; j < 2; ++j) {
            int col = bn + wn + j * 16 + l15;
            float bv = bias[col];
#pragma unroll
            for (int r = 0; r < 8; ++r) {
                int row = bm + wm + i * 16 + r + (hi ? 8 : 0);
                out[(size_t)row * HID_ + col] = f2bf(acc[i][j][r] + bv);
            }
        }
}

// ---------------------------------------------------------------------------
// Kernel 2: per (b,h,t) squared L2 norm of q head vector (monotone for top-k).
// ---------------------------------------------------------------------------
__global__ __launch_bounds__(256) void qnorm_kernel(const __bf16* __restrict__ Q,
                                                    float* __restrict__ norms) {
    int g = blockIdx.x * 256 + threadIdx.x;       // 0 .. B_*NH_*S_-1
    int b = g >> 16;
    int rem = g & 65535;
    int h = rem >> 12;
    int t = rem & 4095;
    const __bf16* p = Q + (size_t)(b * S_ + t) * HID_ + h * HD_;
    float s = 0.f;
#pragma unroll
    for (int d = 0; d < HD_; ++d) { float x = (float)p[d]; s += x * x; }
    norms[(size_t)(b * NH_ + h) * S_ + t] = s;
}

// ---------------------------------------------------------------------------
// Kernel 3: per (b,h) indices = [0, top-62 by norm, 4095] (iterative argmax).
// ---------------------------------------------------------------------------
__global__ __launch_bounds__(256) void topk_kernel(const float* __restrict__ norms,
                                                   int* __restrict__ idxOut) {
    __shared__ float vals[S_];
    __shared__ float rv[256];
    __shared__ int   ri[256];
    const int bh = blockIdx.x, tid = threadIdx.x;
    const float* src = norms + (size_t)bh * S_;
    for (int i = tid; i < S_; i += 256) vals[i] = src[i];
    if (tid == 0) { idxOut[bh * TOPK_] = 0; idxOut[bh * TOPK_ + 63] = S_ - 1; }
    __syncthreads();
    for (int it = 0; it < TOPK_ - 2; ++it) {
        float bm = -1.f; int bi = 0;
        for (int i = tid; i < S_; i += 256) {
            float v = vals[i];
            if (v > bm) { bm = v; bi = i; }
        }
        rv[tid] = bm; ri[tid] = bi;
        __syncthreads();
        for (int s = 128; s > 0; s >>= 1) {
            if (tid < s && rv[tid + s] > rv[tid]) { rv[tid] = rv[tid + s]; ri[tid] = ri[tid + s]; }
            __syncthreads();
        }
        if (tid == 0) { idxOut[bh * TOPK_ + 1 + it] = ri[0]; vals[ri[0]] = -1.f; }
        __syncthreads();
    }
}

// ---------------------------------------------------------------------------
// Kernel 4: block-local attention. 1 block per (b,h,blk); 8 waves x 16 queries.
// Scores (128x256) in registers (16 C-tiles/wave); softmax via half-wave
// shfl reductions; V staged transposed [d][key] in LDS per 64-key chunk;
// P restaged per 32-key step through a per-wave LDS tile -> A frags.
// ---------------------------------------------------------------------------
__global__ __launch_bounds__(256) void local_attn(const __bf16* __restrict__ Q,
                                                  const __bf16* __restrict__ K,
                                                  const __bf16* __restrict__ V,
                                                  float* __restrict__ out) {
    __shared__ __bf16 ldsV[HD_ * 72];        // [d][key(64) pad->72]
    __shared__ __bf16 ldsP[8][16 * 40];      // per-wave [16 q][32 key pad->40]

    const int blkid = blockIdx.x;            // b*512 + h*32 + blk
    const int b = blkid >> 9;
    const int h = (blkid >> 5) & 15;
    const int blk = blkid & 31;
    const int tid = threadIdx.x;
    const int wave = tid >> 5;
    const int lane = tid & 31;
    const int hi = lane >> 4, l15 = lane & 15;
    const int klo = hi ? 8 : 0, klb = hi ? 16 : 0;

    const int q0 = blk * BLK_;
    const int win0 = q0 - BLK_ / 2;
    const size_t ho = (size_t)h * HD_;

    const int qrow = q0 + (wave << 4) + l15;
    const __bf16* qp = Q + (size_t)(b * S_ + qrow) * HID_ + ho;
    v16bf qf[4];
#pragma unroll
    for (int dk = 0; dk < 4; ++dk)
        qf[dk] = load_frag1(qp + dk * 32 + klb);

    // ---- scores: 16 key-subtiles x 4 d-steps of WMMA, K B-frags from global
    v8f s[16];
#pragma unroll
    for (int nk = 0; nk < 16; ++nk) {
        int key = win0 + nk * 16 + l15;
        int keyc = key < 0 ? 0 : (key > S_ - 1 ? S_ - 1 : key);
        const __bf16* kp = K + (size_t)(b * S_ + keyc) * HID_ + ho;
        v8f a = (v8f){0,0,0,0,0,0,0,0};
#pragma unroll
        for (int dk = 0; dk < 4; ++dk)
            a = wmma_bf16(qf[dk], load_frag1(kp + dk * 32 + klb), a);
        bool valid = (key >= 0) && (key < S_);
#pragma unroll
        for (int r = 0; r < 8; ++r) a[r] = valid ? a[r] * SCALE_ : NEGV_;
        s[nk] = a;
    }

    // ---- softmax across 256 keys (16 subtiles x 16 lanes of this half)
#pragma unroll
    for (int r = 0; r < 8; ++r) {
        float m = -3.0e38f;
#pragma unroll
        for (int nk = 0; nk < 16; ++nk) m = fmaxf(m, s[nk][r]);
        m = half16_max(m);
        float t = 0.f;
#pragma unroll
        for (int nk = 0; nk < 16; ++nk) { float e = __expf(s[nk][r] - m); s[nk][r] = e; t += e; }
        t = half16_sum(t);
        float inv = 1.0f / t;
#pragma unroll
        for (int nk = 0; nk < 16; ++nk) s[nk][r] *= inv;
    }

    // ---- ctx = P @ V over 4 chunks of 64 keys
    v8f c[8];
#pragma unroll
    for (int nd = 0; nd < 8; ++nd) c[nd] = (v8f){0,0,0,0,0,0,0,0};

#pragma unroll
    for (int kc = 0; kc < 4; ++kc) {
        __syncthreads();
#pragma unroll
        for (int i = 0; i < 4; ++i) {
            int idx = tid + i * 256;           // 64 keys x 16 d8-groups
            int key = idx >> 4;
            int d8 = (idx & 15) << 3;
            int gk = win0 + kc * 64 + key;
            int gkc = gk < 0 ? 0 : (gk > S_ - 1 ? S_ - 1 : gk);
            const __bf16* vp = V + (size_t)(b * S_ + gkc) * HID_ + ho + d8;
            alignas(16) __bf16 tmp[8];
            *(uint4*)tmp = *(const uint4*)vp;
#pragma unroll
            for (int j = 0; j < 8; ++j) ldsV[(d8 + j) * 72 + key] = tmp[j];
        }
        __syncthreads();

#pragma unroll
        for (int kk = 0; kk < 2; ++kk) {
            int nkb = kc * 4 + kk * 2;
            __bf16* P = ldsP[wave];
#pragma unroll
            for (int r = 0; r < 8; ++r) {
                int m = r + (hi ? 8 : 0);
                P[m * 40 + l15]      = f2bf(s[nkb][r]);
                P[m * 40 + 16 + l15] = f2bf(s[nkb + 1][r]);
            }
            const __bf16* pb = &P[l15 * 40];
            v16bf pf = load_frag1(pb + klb);
#pragma unroll
            for (int nd = 0; nd < 8; ++nd) {
                const __bf16* vb = &ldsV[(nd * 16 + l15) * 72 + kk * 32];
                c[nd] = wmma_bf16(pf, load_frag1(vb + klb), c[nd]);
            }
        }
    }

#pragma unroll
    for (int nd = 0; nd < 8; ++nd) {
        int d = nd * 16 + l15;
#pragma unroll
        for (int r = 0; r < 8; ++r) {
            int t = q0 + (wave << 4) + r + (hi ? 8 : 0);
            out[(size_t)(b * S_ + t) * HID_ + ho + d] = c[nd][r];
        }
    }
}

// ---------------------------------------------------------------------------
// Kernel 5: global attention for the 64 selected queries per (b,h).
// Flash-style online softmax over 32-key chunks; waves 0-3 compute 16 queries
// each; all 8 waves stage V transposed. Runs after local_attn (overwrite =
// reference .set()).
// ---------------------------------------------------------------------------
__global__ __launch_bounds__(256) void global_attn(const __bf16* __restrict__ Q,
                                                   const __bf16* __restrict__ K,
                                                   const __bf16* __restrict__ V,
                                                   const int* __restrict__ idxArr,
                                                   float* __restrict__ out) {
    __shared__ __bf16 ldsV[HD_ * 40];       // [d][32 keys pad->40]
    __shared__ __bf16 ldsP[4][16 * 40];

    const int bh = blockIdx.x;
    const int b = bh >> 4, h = bh & 15;
    const int tid = threadIdx.x;
    const int wave = tid >> 5, lane = tid & 31;
    const int hi = lane >> 4, l15 = lane & 15;
    const int klo = hi ? 8 : 0, klb = hi ? 16 : 0;
    const size_t ho = (size_t)h * HD_;
    const int* myidx = idxArr + bh * TOPK_;
    const bool compute = wave < 4;

    v16bf qf[4];
    if (compute) {
        int tq = myidx[(wave << 4) + l15];
        const __bf16* qp = Q + (size_t)(b * S_ + tq) * HID_ + ho;
#pragma unroll
        for (int dk = 0; dk < 4; ++dk)
            qf[dk] = load_frag1(qp + dk * 32 + klb);
    }

    v8f acc[8];
    float mrun[8], lrun[8];
#pragma unroll
    for (int nd = 0; nd < 8; ++nd) acc[nd] = (v8f){0,0,0,0,0,0,0,0};
#pragma unroll
    for (int r = 0; r < 8; ++r) { mrun[r] = -3.0e38f; lrun[r] = 0.f; }

    for (int kc = 0; kc < S_; kc += 32) {
        __syncthreads();
#pragma unroll
        for (int i = 0; i < 2; ++i) {
            int idx = tid + i * 256;
            int key = idx >> 4;
            int d8 = (idx & 15) << 3;
            const __bf16* vp = V + (size_t)(b * S_ + kc + key) * HID_ + ho + d8;
            alignas(16) __bf16 tmp[8];
            *(uint4*)tmp = *(const uint4*)vp;
#pragma unroll
            for (int j = 0; j < 8; ++j) ldsV[(d8 + j) * 40 + key] = tmp[j];
        }
        __syncthreads();

        if (compute) {
            v8f s0 = (v8f){0,0,0,0,0,0,0,0}, s1 = s0;
            const __bf16* kp0 = K + (size_t)(b * S_ + kc + l15) * HID_ + ho;
            const __bf16* kp1 = K + (size_t)(b * S_ + kc + 16 + l15) * HID_ + ho;
#pragma unroll
            for (int dk = 0; dk < 4; ++dk) {
                s0 = wmma_bf16(qf[dk], load_frag1(kp0 + dk * 32 + klb), s0);
                s1 = wmma_bf16(qf[dk], load_frag1(kp1 + dk * 32 + klb), s1);
            }
#pragma unroll
            for (int r = 0; r < 8; ++r) {
                float a0 = s0[r] * SCALE_, a1 = s1[r] * SCALE_;
                float cm = half16_max(fmaxf(a0, a1));
                float mn = fmaxf(mrun[r], cm);
                float alpha = __expf(mrun[r] - mn);
                float e0 = __expf(a0 - mn), e1 = __expf(a1 - mn);
                float ls = half16_sum(e0 + e1);
                lrun[r] = lrun[r] * alpha + ls;
                mrun[r] = mn;
                s0[r] = e0; s1[r] = e1;
#pragma unroll
                for (int nd = 0; nd < 8; ++nd) acc[nd][r] *= alpha;
            }
            __bf16* P = ldsP[wave];
#pragma unroll
            for (int r = 0; r < 8; ++r) {
                int m = r + (hi ? 8 : 0);
                P[m * 40 + l15]      = f2bf(s0[r]);
                P[m * 40 + 16 + l15] = f2bf(s1[r]);
            }
            const __bf16* pb = &P[l15 * 40];
            v16bf pf = load_frag1(pb + klb);
#pragma unroll
            for (int nd = 0; nd < 8; ++nd) {
                const __bf16* vb = &ldsV[(nd * 16 + l15) * 40];
                acc[nd] = wmma_bf16(pf, load_frag1(vb + klb), acc[nd]);
            }
        }
    }

    if (compute) {
#pragma unroll
        for (int r = 0; r < 8; ++r) {
            int m = r + (hi ? 8 : 0);
            int t = myidx[(wave << 4) + m];
            float inv = 1.0f / lrun[r];
#pragma unroll
            for (int nd = 0; nd < 8; ++nd) {
                int d = nd * 16 + l15;
                out[(size_t)(b * S_ + t) * HID_ + ho + d] = acc[nd][r] * inv;
            }
        }
    }
}

// ---------------------------------------------------------------------------
extern "C" void kernel_launch(void* const* d_in, const int* in_sizes, int n_in,
                              void* d_out, int out_size, void* d_ws, size_t ws_size,
                              hipStream_t stream) {
    const float* H  = (const float*)d_in[0];
    const float* Wq = (const float*)d_in[1];
    const float* bq = (const float*)d_in[2];
    const float* Wk = (const float*)d_in[3];
    const float* bk = (const float*)d_in[4];
    const float* Wv = (const float*)d_in[5];
    const float* bv = (const float*)d_in[6];
    float* out = (float*)d_out;

    const size_t QKV = (size_t)B_ * S_ * HID_;      // 16.78M elems
    const size_t WSZ = (size_t)HID_ * HID_;         // 4.19M elems
    __bf16* Qbf = (__bf16*)d_ws;
    __bf16* Kbf = Qbf + QKV;
    __bf16* Vbf = Kbf + QKV;
    __bf16* Hbf = Vbf + QKV;
    __bf16* Wt  = Hbf + QKV;                        // reused per projection
    float* norms = (float*)(Wt + WSZ);              // B_*NH_*S_ floats
    int* idxArr = (int*)(norms + (size_t)B_ * NH_ * S_);

    dim3 cwGrid(HID_ / 64, HID_ / 64);              // 32 x 32
    dim3 gemmGrid(HID_ / 128, (B_ * S_) / 128);     // 16 x 64

    cvt_h<<<(int)(QKV / (256 * 8)), 256, 0, stream>>>(H, Hbf);

    cvt_w<<<cwGrid, 256, 0, stream>>>(Wq, Wt);
    qkv_gemm<<<gemmGrid, 256, 0, stream>>>(Hbf, Wt, bq, Qbf);
    cvt_w<<<cwGrid, 256, 0, stream>>>(Wk, Wt);
    qkv_gemm<<<gemmGrid, 256, 0, stream>>>(Hbf, Wt, bk, Kbf);
    cvt_w<<<cwGrid, 256, 0, stream>>>(Wv, Wt);
    qkv_gemm<<<gemmGrid, 256, 0, stream>>>(Hbf, Wt, bv, Vbf);

    qnorm_kernel<<<(B_ * NH_ * S_) / 256, 256, 0, stream>>>(Qbf, norms);
    topk_kernel<<<B_ * NH_, 256, 0, stream>>>(norms, idxArr);

    local_attn<<<B_ * NH_ * NB_, 256, 0, stream>>>(Qbf, Kbf, Vbf, out);
    global_attn<<<B_ * NH_, 256, 0, stream>>>(Qbf, Kbf, Vbf, idxArr, out);
}
